// Generator_75385265979727
// MI455X (gfx1250) — matrix operand order repaired and checked by reading, not compiled, not run on hardware
//
#include <hip/hip_runtime.h>
#include <hip/hip_bf16.h>
#include <math.h>
#include <stdint.h>

typedef __attribute__((ext_vector_type(16))) _Float16 v16h;
typedef __attribute__((ext_vector_type(8)))  float    v8f;

#define B_    4
#define W_    12
#define N_    1024
#define FIN_  32
#define FOUT_ 32
#define H_    256
#define D_    (N_*FOUT_)   /* 32768 */
#define G4_   (4*H_)       /* 1024  */

// ---------------------------------------------------------------------------
// 1) degree: d[bw][n] = rsqrt(1 + sum_m S[bw][n][m])   (A = S + I self loop)
// ---------------------------------------------------------------------------
__global__ void k_degree(const float* __restrict__ S, float* __restrict__ dbuf) {
  int lane = threadIdx.x & 31;
  int wave = threadIdx.x >> 5;
  int row  = blockIdx.x * 8 + wave;                 // row in [0, B*W*N)
  const float* r = S + (size_t)row * N_;
  float s = 0.f;
  for (int i = lane; i < N_; i += 32) s += r[i];
#pragma unroll
  for (int o = 16; o > 0; o >>= 1) s += __shfl_xor(s, o, 32);
  if (lane == 0) dbuf[row] = rsqrtf(s + 1.0f);
}

// ---------------------------------------------------------------------------
// 2) Vt[bw][f][n] = (fp16) d[bw][n] * (in_node[bw][n][:] . gcn_w[w][:][f])
// ---------------------------------------------------------------------------
__global__ void k_vt(const float* __restrict__ node, const float* __restrict__ gw,
                     const float* __restrict__ dbuf, _Float16* __restrict__ Vt) {
  __shared__ float wsh[FIN_ * FOUT_];
  int bw    = blockIdx.x >> 7;                      // 128 blocks per (b,w)
  int w     = bw % W_;
  int local = ((blockIdx.x & 127) << 8) + threadIdx.x;  // 0..32767
  int n     = local >> 5;
  int f     = local & 31;
  for (int i = threadIdx.x; i < FIN_ * FOUT_; i += 256)
    wsh[i] = gw[(size_t)w * FIN_ * FOUT_ + i];
  __syncthreads();
  const float* xr = node + ((size_t)bw * N_ + n) * FIN_;
  float acc = 0.f;
#pragma unroll
  for (int k = 0; k < FIN_; ++k) acc += xr[k] * wsh[k * FOUT_ + f];
  float dn = dbuf[(size_t)bw * N_ + n];
  Vt[((size_t)bw * FOUT_ + f) * N_ + n] = (_Float16)(dn * acc);
}

// ---------------------------------------------------------------------------
// 3) WMMA GEMM: Y = S[1024x1024] @ V[1024x32] per (b,w); then
//    x[w][b][row*32+col] = d[row] * (Y[row,col] + V[row,col])
// The 64KB Vt[bw] tile is staged once per block into LDS with the CDNA5
// async global->LDS path (ASYNCcnt), then B-operands come from LDS while the
// in_shots A-stream flows through VMEM into two v_wmma per K step.
// ---------------------------------------------------------------------------
__global__ void __launch_bounds__(256) k_gcn(const float* __restrict__ S,
                                             const _Float16* __restrict__ Vt,
                                             const float* __restrict__ dbuf,
                                             float* __restrict__ x) {
  __shared__ _Float16 shv[FOUT_ * N_];              // 64 KB (of 320 KB/WGP)
  int lane  = threadIdx.x & 31;
  int wave  = threadIdx.x >> 5;
  int bw    = blockIdx.x >> 3;
  int b     = bw / W_;
  int w     = bw % W_;
  int mtile = ((blockIdx.x & 7) << 3) + wave;       // 64 m-tiles per (b,w)
  int m0    = mtile << 4;
  int half  = lane >> 4;                            // lane group 0/1
  int lr    = lane & 15;

  const float*    Sbw  = S  + (size_t)bw * N_ * N_;
  const _Float16* Vtbw = Vt + (size_t)bw * FOUT_ * N_;

  // --- async copy Vt[bw] (65536 B = 4096 x b128) into LDS ---
  {
    const char* src  = (const char*)Vtbw;
    unsigned    lds0 = (unsigned)(uintptr_t)shv;
#pragma unroll
    for (int it = 0; it < 16; ++it) {
      int      bofs = (it * 256 + threadIdx.x) * 16;
      unsigned lds  = lds0 + (unsigned)bofs;
      const void* gp = (const void*)(src + bofs);
      asm volatile("global_load_async_to_lds_b128 %0, %1, off"
                   :: "v"(lds), "v"(gp) : "memory");
    }
    asm volatile("s_wait_asynccnt 0x0" ::: "memory");
    __syncthreads();
  }

  const float*    Srow = Sbw + (size_t)(m0 + lr) * N_;
  const _Float16* sc0  = shv + (size_t)lr * N_;          // col lr      (n-tile 0)
  const _Float16* sc1  = shv + (size_t)(16 + lr) * N_;   // col 16+lr   (n-tile 1)

  v8f acc0 = {}; v8f acc1 = {};
  for (int kt = 0; kt < N_; kt += 32) {
    // A 16x32 f16, k order this code uses (same in B): lane group 0 -> K kt..kt+15 of row lr,
    //                     lane group 1 -> same rows, K kt+16..kt+31.
    int ka = kt + half * 16;
    int kb = kt + half * 16 + 8;
    v16h a;
#pragma unroll
    for (int i = 0; i < 8; ++i) {
      a[i]     = (_Float16)Srow[ka + i];
      a[8 + i] = (_Float16)Srow[kb + i];
    }
    __builtin_prefetch(Srow + kt + 64, 0, 0);       // global_prefetch_b8
    // B 32x16 f16, same k order as A: lane group 0 -> K kt..kt+15, group 1 -> kt+16..kt+31
    int kc = kt + half * 16;
    v16h b0, b1;
#pragma unroll
    for (int i = 0; i < 16; ++i) {
      b0[i] = sc0[kc + i];
      b1[i] = sc1[kc + i];
    }
    acc0 = __builtin_amdgcn_wmma_f32_16x16x32_f16(false, a, false, b0, (short)0, acc0, false, false);
    acc1 = __builtin_amdgcn_wmma_f32_16x16x32_f16(false, a, false, b1, (short)0, acc1, false, false);
  }

  // C/D layout: VGPR v -> (M = v + 8*laneGroup, N = lr)
  float* xbw = x + ((size_t)w * B_ + b) * D_;       // time-major [W][B][D]
#pragma unroll
  for (int v = 0; v < 8; ++v) {
    int   row = m0 + v + half * 8;
    float dr  = dbuf[(size_t)bw * N_ + row];
    float y0  = acc0[v] + (float)sc0[row];          // + self-loop term V[row,col]
    float y1  = acc1[v] + (float)sc1[row];
    xbw[(size_t)row * FOUT_ + lr]      = dr * y0;
    xbw[(size_t)row * FOUT_ + 16 + lr] = dr * y1;
  }
}

// ---------------------------------------------------------------------------
// 4) LSTM: init, per-step gates, per-step cell update
// One block per gate output j; 4 batch accumulators so each w_ih row is
// streamed exactly once per step (L2-resident across the 12 steps).
// ---------------------------------------------------------------------------
__global__ void k_init_hc(float* __restrict__ h, float* __restrict__ c) {
  int i = blockIdx.x * 256 + threadIdx.x;
  if (i < B_ * H_) { h[i] = 0.f; c[i] = 0.f; }
}

__global__ void k_gates(const float* __restrict__ x, const float* __restrict__ wih,
                        const float* __restrict__ whh, const float* __restrict__ bih,
                        const float* __restrict__ bhh, const float* __restrict__ h,
                        float* __restrict__ g, int t) {
  __shared__ float red[8][4];
  int tid = threadIdx.x;
  int j   = blockIdx.x;                             // gate output in [0, 4H)
  const float* wr = wih + (size_t)j * D_;
  const float* x0 = x + (size_t)t * B_ * D_;        // batch rows x0 + b*D_
  float a0 = 0.f, a1 = 0.f, a2 = 0.f, a3 = 0.f;
  for (int k = tid * 4; k < D_; k += 1024) {
    float4 wv = *(const float4*)(wr + k);
    float4 v0 = *(const float4*)(x0 + k);
    float4 v1 = *(const float4*)(x0 + D_ + k);
    float4 v2 = *(const float4*)(x0 + 2 * D_ + k);
    float4 v3 = *(const float4*)(x0 + 3 * D_ + k);
    a0 += v0.x * wv.x + v0.y * wv.y + v0.z * wv.z + v0.w * wv.w;
    a1 += v1.x * wv.x + v1.y * wv.y + v1.z * wv.z + v1.w * wv.w;
    a2 += v2.x * wv.x + v2.y * wv.y + v2.z * wv.z + v2.w * wv.w;
    a3 += v3.x * wv.x + v3.y * wv.y + v3.z * wv.z + v3.w * wv.w;
  }
  float wh = whh[(size_t)j * H_ + tid];             // H_ == blockDim
  a0 += h[0 * H_ + tid] * wh;
  a1 += h[1 * H_ + tid] * wh;
  a2 += h[2 * H_ + tid] * wh;
  a3 += h[3 * H_ + tid] * wh;
#pragma unroll
  for (int o = 16; o > 0; o >>= 1) {
    a0 += __shfl_xor(a0, o, 32);
    a1 += __shfl_xor(a1, o, 32);
    a2 += __shfl_xor(a2, o, 32);
    a3 += __shfl_xor(a3, o, 32);
  }
  if ((tid & 31) == 0) {
    int wv = tid >> 5;
    red[wv][0] = a0; red[wv][1] = a1; red[wv][2] = a2; red[wv][3] = a3;
  }
  __syncthreads();
  if (tid < 4) {
    float tot = 0.f;
#pragma unroll
    for (int i = 0; i < 8; ++i) tot += red[i][tid];
    g[tid * G4_ + j] = tot + bih[j] + bhh[j];
  }
}

__global__ void k_cell(const float* __restrict__ g, float* __restrict__ h,
                       float* __restrict__ c) {
  int b = blockIdx.x;
  int j = threadIdx.x;
  const float* gb = g + b * G4_;
  float gi = gb[j], gf = gb[H_ + j], gg = gb[2 * H_ + j], go = gb[3 * H_ + j];
  float si = 1.f / (1.f + __expf(-gi));
  float sf = 1.f / (1.f + __expf(-gf));
  float so = 1.f / (1.f + __expf(-go));
  float cn = sf * c[b * H_ + j] + si * tanhf(gg);
  c[b * H_ + j] = cn;
  h[b * H_ + j] = so * tanhf(cn);
}

// ---------------------------------------------------------------------------
// 5) FC1:  out[b][m] = hn[b][:] . f1_w[m][:] + f1_b[m]
// ---------------------------------------------------------------------------
__global__ void k_fc1(const float* __restrict__ h, const float* __restrict__ f1w,
                      const float* __restrict__ f1b, float* __restrict__ out) {
  __shared__ float sh[H_];
  int b = blockIdx.x >> 7;                          // 128 blocks per batch
  int m = ((blockIdx.x & 127) << 8) + threadIdx.x;
  sh[threadIdx.x] = h[b * H_ + threadIdx.x];        // blockDim == H_
  __syncthreads();
  const float* wr = f1w + (size_t)m * H_;
  float acc = f1b[m];
  for (int k = 0; k < H_; ++k) acc += sh[k] * wr[k];
  out[(size_t)b * D_ + m] = acc;
}

// ---------------------------------------------------------------------------
// 6) s1[b][n] = out[b][n][:] . f2_w[:32] ; s2 with f2_w[32:]
// ---------------------------------------------------------------------------
__global__ void k_scores(const float* __restrict__ out, const float* __restrict__ f2w,
                         float* __restrict__ s1, float* __restrict__ s2) {
  int idx = blockIdx.x * 256 + threadIdx.x;         // over B*N
  int b = idx >> 10, n = idx & 1023;
  const float* o = out + (size_t)b * D_ + n * FOUT_;
  float a = 0.f, c = 0.f;
#pragma unroll
  for (int f = 0; f < FOUT_; ++f) {
    a += o[f] * f2w[f];
    c += o[f] * f2w[FOUT_ + f];
  }
  s1[idx] = a; s2[idx] = c;
}

// ---------------------------------------------------------------------------
// 7) pairwise compare_and_update + L1 row normalize (one block per (b,i) row)
// ---------------------------------------------------------------------------
__global__ void k_final(const float* __restrict__ s1, const float* __restrict__ s2,
                        const float* __restrict__ f2b, float* __restrict__ res) {
  __shared__ float red[8];
  int b   = blockIdx.x >> 10;
  int i   = blockIdx.x & 1023;
  int tid = threadIdx.x;
  float bias = f2b[0];
  float s1i = s1[b * N_ + i];
  float s2i = s2[b * N_ + i];
  float vals[4];
  float asum = 0.f;
#pragma unroll
  for (int q = 0; q < 4; ++q) {
    int   j   = tid + q * 256;
    float sij = s1i + s2[b * N_ + j] + bias;
    float sji = s1[b * N_ + j] + s2i + bias;
    float r;
    if (i == j)      r = sij;
    else if (i < j)  r = ((sij >= sji) && (sij > 0.f))  ? sij : 0.f;
    else             r = ((sij >  sji) && (sij >= 0.f)) ? sij : 0.f;
    vals[q] = r;
    asum += fabsf(r);
  }
#pragma unroll
  for (int o = 16; o > 0; o >>= 1) asum += __shfl_xor(asum, o, 32);
  if ((tid & 31) == 0) red[tid >> 5] = asum;
  __syncthreads();
  float l1 = 0.f;
#pragma unroll
  for (int q = 0; q < 8; ++q) l1 += red[q];
  float inv = 1.f / fmaxf(l1, 1e-12f);
  float* rrow = res + (size_t)(b * N_ + i) * N_;
#pragma unroll
  for (int q = 0; q < 4; ++q) rrow[tid + q * 256] = vals[q] * inv;
}

// ---------------------------------------------------------------------------
extern "C" void kernel_launch(void* const* d_in, const int* in_sizes, int n_in,
                              void* d_out, int out_size, void* d_ws, size_t ws_size,
                              hipStream_t stream) {
  const float* in_shots = (const float*)d_in[0];
  const float* in_node  = (const float*)d_in[1];
  // d_in[2] (a) unused by reference
  const float* gcn_w    = (const float*)d_in[3];
  const float* w_ih     = (const float*)d_in[4];
  const float* w_hh     = (const float*)d_in[5];
  const float* b_ih     = (const float*)d_in[6];
  const float* b_hh     = (const float*)d_in[7];
  const float* f1_w     = (const float*)d_in[8];
  const float* f1_b     = (const float*)d_in[9];
  const float* f2_w     = (const float*)d_in[10];
  const float* f2_b     = (const float*)d_in[11];
  float* resOut = (float*)d_out;

  // workspace layout (256B aligned, ~10.2 MB total)
  char* ws = (char*)d_ws;
  size_t off = 0;
  auto take = [&](size_t bytes) -> char* {
    char* p = ws + off;
    off += (bytes + 255) & ~(size_t)255;
    return p;
  };
  float*    dbuf = (float*)   take((size_t)B_ * W_ * N_ * 4);        // 192 KB
  _Float16* Vt   = (_Float16*)take((size_t)B_ * W_ * FOUT_ * N_ * 2);// 3 MB
  float*    xbuf = (float*)   take((size_t)W_ * B_ * D_ * 4);        // 6 MB
  float*    gbuf = (float*)   take((size_t)B_ * G4_ * 4);
  float*    hbuf = (float*)   take((size_t)B_ * H_ * 4);
  float*    cbuf = (float*)   take((size_t)B_ * H_ * 4);
  float*    outb = (float*)   take((size_t)B_ * D_ * 4);
  float*    s1   = (float*)   take((size_t)B_ * N_ * 4);
  float*    s2   = (float*)   take((size_t)B_ * N_ * 4);
  (void)ws_size; (void)in_sizes; (void)n_in; (void)out_size;

  // GCN front end
  k_degree<<<dim3(B_ * W_ * N_ / 8), dim3(256), 0, stream>>>(in_shots, dbuf);
  k_vt    <<<dim3(B_ * W_ * 128),    dim3(256), 0, stream>>>(in_node, gcn_w, dbuf, Vt);
  k_gcn   <<<dim3(B_ * W_ * 8),      dim3(256), 0, stream>>>(in_shots, Vt, dbuf, xbuf);

  // LSTM over W time steps
  k_init_hc<<<dim3((B_ * H_ + 255) / 256), dim3(256), 0, stream>>>(hbuf, cbuf);
  for (int t = 0; t < W_; ++t) {
    k_gates<<<dim3(G4_), dim3(256), 0, stream>>>(xbuf, w_ih, w_hh, b_ih, b_hh, hbuf, gbuf, t);
    k_cell <<<dim3(B_),  dim3(H_),  0, stream>>>(gbuf, hbuf, cbuf);
  }

  // head
  k_fc1   <<<dim3(B_ * 128), dim3(256), 0, stream>>>(hbuf, f1_w, f1_b, outb);
  k_scores<<<dim3(B_ * N_ / 256), dim3(256), 0, stream>>>(outb, f2_w, s1, s2);
  k_final <<<dim3(B_ * N_), dim3(256), 0, stream>>>(s1, s2, f2_b, resOut);
}
